// CustomAutoencoder_59030030516941
// MI455X (gfx1250) — compile-verified, matrix operand order checked by reading, not run on hardware
//
#include <hip/hip_runtime.h>
#include <hip/hip_bf16.h>
#include <math.h>

typedef __attribute__((ext_vector_type(16))) _Float16 v16h;
typedef __attribute__((ext_vector_type(8)))  float    v8f;

// ---------------------------------------------------------------------------
// Load 16 consecutive f32 values, convert to an f16 WMMA fragment slice.
// 16x16x32 f16 WMMA: A-frag lane = row M=(lane&15), K-run kb..kb+15 with
// element e <-> K=kb+e; B-frag lane = col N=(lane&15), same K packing.
// ---------------------------------------------------------------------------
__device__ __forceinline__ v16h load16_f16(const float* __restrict__ p) {
  v16h f;
  const float4* q = reinterpret_cast<const float4*>(p);
#pragma unroll
  for (int i = 0; i < 4; ++i) {
    float4 t = q[i];
    f[4 * i + 0] = (_Float16)t.x;
    f[4 * i + 1] = (_Float16)t.y;
    f[4 * i + 2] = (_Float16)t.z;
    f[4 * i + 3] = (_Float16)t.w;
  }
  return f;
}

__device__ __forceinline__ v16h zero16() {
  v16h f;
#pragma unroll
  for (int i = 0; i < 16; ++i) f[i] = (_Float16)0.0f;
  return f;
}

// ---------------------------------------------------------------------------
// FC GEMM: C[M,N] = act(A[M,K] @ W[N,K]^T + bias), one wave per 16x16 tile.
// ACT: 0=none, 1=relu, 2=leaky_relu(0.01)
// ---------------------------------------------------------------------------
template <int ACT>
__global__ void gemm_wmma(const float* __restrict__ A, const float* __restrict__ W,
                          const float* __restrict__ bias, float* __restrict__ C,
                          int M, int N, int K) {
  const int ntn  = N >> 4;
  const int tile = blockIdx.x;
  const int tm   = tile / ntn;
  const int tn   = tile - tm * ntn;
  const int lane = threadIdx.x & 31;
  const int grp  = lane >> 4;
  const int mn   = lane & 15;

  const int arow = tm * 16 + mn;
  const int bcol = tn * 16 + mn;

  v8f acc = {};
  for (int k0 = 0; k0 < K; k0 += 32) {
    const int kb = k0 + grp * 16;
    v16h a = load16_f16(A + (size_t)arow * K + kb);
    v16h b = load16_f16(W + (size_t)bcol * K + kb);
    acc = __builtin_amdgcn_wmma_f32_16x16x32_f16(false, a, false, b,
                                                 (short)0, acc, false, false);
  }

  const float bv = bias[bcol];
#pragma unroll
  for (int v = 0; v < 8; ++v) {
    const int row = tm * 16 + v + 8 * grp;
    float r = acc[v] + bv;
    if (ACT == 1) r = fmaxf(r, 0.0f);
    if (ACT == 2) r = (r > 0.0f) ? r : 0.01f * r;
    C[(size_t)row * N + bcol] = r;
  }
}

// ---------------------------------------------------------------------------
// CustomLayer: relu( (A@Wc^T + bc) + tanh(tanh(low*(A@Wn^T + bn) - high)) )
// Two accumulators share the A fragment; low/high/bias are per-lane scalars.
// ---------------------------------------------------------------------------
__global__ void custom_layer_wmma(const float* __restrict__ A,
                                  const float* __restrict__ Wc, const float* __restrict__ bc,
                                  const float* __restrict__ Wn, const float* __restrict__ bn,
                                  const float* __restrict__ low, const float* __restrict__ high,
                                  float* __restrict__ C, int M, int N, int K) {
  const int ntn  = N >> 4;
  const int tile = blockIdx.x;
  const int tm   = tile / ntn;
  const int tn   = tile - tm * ntn;
  const int lane = threadIdx.x & 31;
  const int grp  = lane >> 4;
  const int mn   = lane & 15;

  const int arow = tm * 16 + mn;
  const int bcol = tn * 16 + mn;

  v8f accc = {};
  v8f accn = {};
  for (int k0 = 0; k0 < K; k0 += 32) {
    const int kb = k0 + grp * 16;
    v16h a  = load16_f16(A + (size_t)arow * K + kb);
    v16h wc = load16_f16(Wc + (size_t)bcol * K + kb);
    v16h wn = load16_f16(Wn + (size_t)bcol * K + kb);
    accc = __builtin_amdgcn_wmma_f32_16x16x32_f16(false, a, false, wc,
                                                  (short)0, accc, false, false);
    accn = __builtin_amdgcn_wmma_f32_16x16x32_f16(false, a, false, wn,
                                                  (short)0, accn, false, false);
  }

  const float bcv = bc[bcol];
  const float bnv = bn[bcol];
  const float lo  = low[bcol];
  const float hi  = high[bcol];
#pragma unroll
  for (int v = 0; v < 8; ++v) {
    const int row   = tm * 16 + v + 8 * grp;
    const float lin = accc[v] + bcv;
    const float pre = accn[v] + bnv;
    const float nr  = tanhf(tanhf(lo * pre - hi));
    C[(size_t)row * N + bcol] = fmaxf(lin + nr, 0.0f);
  }
}

// ---------------------------------------------------------------------------
// Direct Conv2d, stride 2, pad 1, k=3, fused ReLU. NCHW in/out, OIHW weights.
// Tiny channel counts -> bandwidth-bound; f32 VALU is right here.
// ---------------------------------------------------------------------------
__global__ void conv2d_s2_relu(const float* __restrict__ x, const float* __restrict__ w,
                               const float* __restrict__ bias, float* __restrict__ y,
                               int B, int Cin, int Hin, int Win,
                               int Cout, int Hout, int Wout) {
  const long total = (long)B * Cout * Hout * Wout;
  long idx = (long)blockIdx.x * blockDim.x + threadIdx.x;
  if (idx >= total) return;
  int  ox = (int)(idx % Wout); long t = idx / Wout;
  int  oy = (int)(t % Hout);   t /= Hout;
  int  oc = (int)(t % Cout);
  int  b  = (int)(t / Cout);

  float sum = bias[oc];
  for (int ic = 0; ic < Cin; ++ic) {
    const float* xp = x + ((size_t)(b * Cin + ic) * Hin) * Win;
    const float* wp = w + ((size_t)(oc * Cin + ic) * 3) * 3;
#pragma unroll
    for (int ky = 0; ky < 3; ++ky) {
      const int iy = oy * 2 + ky - 1;
      if ((unsigned)iy >= (unsigned)Hin) continue;
#pragma unroll
      for (int kx = 0; kx < 3; ++kx) {
        const int ix = ox * 2 + kx - 1;
        if ((unsigned)ix >= (unsigned)Win) continue;
        sum = fmaf(xp[(size_t)iy * Win + ix], wp[ky * 3 + kx], sum);
      }
    }
  }
  y[idx] = fmaxf(sum, 0.0f);
}

// ---------------------------------------------------------------------------
// Direct ConvTranspose2d (stride2,pad1,outpad1,k3), NCHW input, torch weight
// [Cin,Cout,3,3]. OUT_NHWC=1 writes NHWC (feeds the WMMA deconvs).
// ---------------------------------------------------------------------------
template <int OUT_NHWC>
__global__ void deconv2d_s2(const float* __restrict__ x, const float* __restrict__ w,
                            const float* __restrict__ bias, float* __restrict__ y,
                            int B, int Cin, int Hin, int Win,
                            int Cout, int Hout, int Wout) {
  const long total = (long)B * Cout * Hout * Wout;
  long idx = (long)blockIdx.x * blockDim.x + threadIdx.x;
  if (idx >= total) return;
  int  ox = (int)(idx % Wout); long t = idx / Wout;
  int  oy = (int)(t % Hout);   t /= Hout;
  int  oc = (int)(t % Cout);
  int  b  = (int)(t / Cout);

  float sum = bias[oc];
#pragma unroll
  for (int ky = 0; ky < 3; ++ky) {
    const int ty = oy + 1 - ky;
    if (ty & 1) continue;
    const int iy = ty >> 1;
    if ((unsigned)iy >= (unsigned)Hin) continue;
#pragma unroll
    for (int kx = 0; kx < 3; ++kx) {
      const int tx = ox + 1 - kx;
      if (tx & 1) continue;
      const int ix = tx >> 1;
      if ((unsigned)ix >= (unsigned)Win) continue;
      for (int ic = 0; ic < Cin; ++ic) {
        const float xv = x[((size_t)(b * Cin + ic) * Hin + iy) * Win + ix];
        const float wv = w[((size_t)(ic * Cout + oc) * 3 + ky) * 3 + kx];
        sum = fmaf(xv, wv, sum);
      }
    }
  }
  if (OUT_NHWC)
    y[(((size_t)b * Hout + oy) * Wout + ox) * Cout + oc] = sum;
  else
    y[idx] = sum;
}

// ---------------------------------------------------------------------------
// Pack transposed-conv weights for implicit GEMM. Parity class c=(py<<1)|px,
// taps (dy,dx) in {0..py}x{0..px}, ky=py+1-2dy, kx=px+1-2dx.
// Layout: Wpack[seg(c) + oc*Kpad(c) + tap*Cin + ic], zero-padded to Kpad
// (K rounded up to 32) and to Cout_pad rows.
// ---------------------------------------------------------------------------
__global__ void pack_deconv_w(const float* __restrict__ w, float* __restrict__ wpack,
                              int Cin, int Cout, int Cout_pad) {
  int Kpad_[4], seg_[4], run = 0;
#pragma unroll
  for (int c = 0; c < 4; ++c) {
    const int py = c >> 1, px = c & 1;
    const int Kp = (py + 1) * (px + 1) * Cin;
    Kpad_[c] = (Kp + 31) & ~31;
    seg_[c]  = Cout_pad * run;
    run += Kpad_[c];
  }
  const int total = Cout_pad * run;
  int idx = blockIdx.x * blockDim.x + threadIdx.x;
  if (idx >= total) return;

  int c = 3;
#pragma unroll
  for (int cc = 0; cc < 4; ++cc)
    if (idx < seg_[cc] + Cout_pad * Kpad_[cc]) { c = cc; break; }
  const int rel = idx - seg_[c];
  const int oc  = rel / Kpad_[c];
  const int kp  = rel - oc * Kpad_[c];
  const int py = c >> 1, px = c & 1;
  const int Kp = (py + 1) * (px + 1) * Cin;

  float v = 0.0f;
  if (kp < Kp && oc < Cout) {
    const int tap = kp / Cin, ic = kp - tap * Cin;
    const int dy = tap / (px + 1), dx = tap - dy * (px + 1);
    const int ky = py + 1 - 2 * dy, kx = px + 1 - 2 * dx;
    v = w[((size_t)(ic * Cout + oc) * 3 + ky) * 3 + kx];
  }
  wpack[idx] = v;
}

// ---------------------------------------------------------------------------
// WMMA implicit-GEMM transposed conv (stride2,pad1,outpad1,k3).
// x: NHWC [B,Hin,Win,Cin] (Cin multiple of 16). One wave per
// (parity class, 16-position tile, 16-outchannel tile).
// OUT_NCHW=1: write NCHW with oc<Cout mask (final layer); else NHWC.
// ---------------------------------------------------------------------------
template <int OUT_NCHW>
__global__ void deconv_wmma(const float* __restrict__ x, const float* __restrict__ wpack,
                            const float* __restrict__ bias, float* __restrict__ y,
                            int B, int Cin, int Hin, int Win, int Cout, int Cout_pad) {
  const int ntm = (B * Hin * Win) >> 4;
  const int ntn = Cout_pad >> 4;
  const int tilesPerClass = ntm * ntn;
  const int c  = blockIdx.x / tilesPerClass;
  const int r  = blockIdx.x - c * tilesPerClass;
  const int tm = r / ntn;
  const int tn = r - tm * ntn;

  const int py = c >> 1, px = c & 1;
  const int Kp   = (py + 1) * (px + 1) * Cin;
  const int Kpad = (Kp + 31) & ~31;
  int seg = 0;
  for (int cc = 0; cc < c; ++cc) {
    const int p2 = cc >> 1, p3 = cc & 1;
    seg += Cout_pad * ((((p2 + 1) * (p3 + 1) * Cin) + 31) & ~31);
  }

  const int lane = threadIdx.x & 31;
  const int grp  = lane >> 4;
  const int mn   = lane & 15;

  // A-row position (b, iy, ix)
  const int pos = tm * 16 + mn;
  const int ix  = pos % Win;
  const int t2  = pos / Win;
  const int iy  = t2 % Hin;
  const int b   = t2 / Hin;

  const int bcol = tn * 16 + mn;
  const float* wrow = wpack + seg + (size_t)bcol * Kpad;

  v8f acc = {};
  for (int k0 = 0; k0 < Kpad; k0 += 32) {
    const int kb = k0 + grp * 16;
    v16h a;
    bool valid = false;
    size_t xoff = 0;
    if (kb < Kp) {
      const int tap  = kb / Cin;
      const int koff = kb - tap * Cin;
      const int dy = tap / (px + 1), dx = tap - (tap / (px + 1)) * (px + 1);
      const int iy2 = iy + dy, ix2 = ix + dx;
      if (iy2 < Hin && ix2 < Win) {
        valid = true;
        xoff = (((size_t)b * Hin + iy2) * Win + ix2) * Cin + koff;
      }
    }
    a = valid ? load16_f16(x + xoff) : zero16();
    v16h bf = load16_f16(wrow + kb);
    acc = __builtin_amdgcn_wmma_f32_16x16x32_f16(false, a, false, bf,
                                                 (short)0, acc, false, false);
  }

  const int  Hout = 2 * Hin, Wout = 2 * Win;
  const bool ocok = (bcol < Cout);
  const float bv  = ocok ? bias[bcol] : 0.0f;
#pragma unroll
  for (int v = 0; v < 8; ++v) {
    const int pos2 = tm * 16 + v + 8 * grp;
    const int ix0  = pos2 % Win;
    const int t3   = pos2 / Win;
    const int iy0  = t3 % Hin;
    const int b0   = t3 / Hin;
    const int oy = 2 * iy0 + py, ox = 2 * ix0 + px;
    const float rv = acc[v] + bv;
    if (OUT_NCHW) {
      if (ocok) y[(((size_t)b0 * Cout + bcol) * Hout + oy) * Wout + ox] = rv;
    } else {
      y[(((size_t)b0 * Hout + oy) * Wout + ox) * Cout + bcol] = rv;
    }
  }
}

// ---------------------------------------------------------------------------
// Host-side launch
// ---------------------------------------------------------------------------
extern "C" void kernel_launch(void* const* d_in, const int* in_sizes, int n_in,
                              void* d_out, int out_size, void* d_ws, size_t ws_size,
                              hipStream_t stream) {
  (void)in_sizes; (void)n_in; (void)out_size; (void)ws_size;

  const float* x       = (const float*)d_in[0];
  const float* conv1_w = (const float*)d_in[1];  const float* conv1_b = (const float*)d_in[2];
  const float* conv2_w = (const float*)d_in[3];  const float* conv2_b = (const float*)d_in[4];
  const float* conv3_w = (const float*)d_in[5];  const float* conv3_b = (const float*)d_in[6];
  const float* conv4_w = (const float*)d_in[7];  const float* conv4_b = (const float*)d_in[8];
  const float* l2_w    = (const float*)d_in[9];  const float* l2_b    = (const float*)d_in[10];
  const float* cl_w    = (const float*)d_in[11]; const float* cl_b    = (const float*)d_in[12];
  const float* n_w     = (const float*)d_in[13]; const float* n_b     = (const float*)d_in[14];
  const float* low     = (const float*)d_in[15]; const float* high    = (const float*)d_in[16];
  const float* l4_w    = (const float*)d_in[17]; const float* l4_b    = (const float*)d_in[18];
  const float* lL_w    = (const float*)d_in[19]; const float* lL_b    = (const float*)d_in[20];
  const float* fc4_w   = (const float*)d_in[21]; const float* fc4_b   = (const float*)d_in[22];
  const float* fc5_w   = (const float*)d_in[23]; const float* fc5_b   = (const float*)d_in[24];
  const float* dc1_w   = (const float*)d_in[25]; const float* dc1_b   = (const float*)d_in[26];
  const float* dc2_w   = (const float*)d_in[27]; const float* dc2_b   = (const float*)d_in[28];
  const float* dc3_w   = (const float*)d_in[29]; const float* dc3_b   = (const float*)d_in[30];
  const float* dc4_w   = (const float*)d_in[31]; const float* dc4_b   = (const float*)d_in[32];

  float* ws  = (float*)d_ws;
  float* out = (float*)d_out;
  const int B = 256;

  // Workspace layout (float offsets). Decoder overlays the dead encoder region.
  float* a1  = ws + 0;            // [256,16,64,64] NCHW    16,777,216
  float* a2  = ws + 16777216;     // [256, 8,32,32] NCHW     2,097,152
  float* a3  = ws + 18874368;     // [256, 4,16,16] NCHW       262,144
  float* a4  = ws + 19136512;     // [256,128] latent           32,768
  float* h1  = ws + 19169280;     // [256,256]                  65,536
  float* h2  = ws + 19234816;     // [256,512]                 131,072
  float* h3  = ws + 19365888;     // [256,256]                  65,536
  float* h4  = ws + 19431424;     // [256,256]                  65,536
  float* h5  = ws + 19496960;     // [256,512]                 131,072
  float* h6  = ws + 19628032;     // [256,512]=[256,8,8,8]     131,072
  float* e1  = ws + 0;            // [256,16,16,16ch] NHWC   1,048,576 (a1 dead)
  float* e2  = ws + 1048576;      // [256,32,32,32ch] NHWC   8,388,608
  float* e3  = ws + 9437184;      // [256,64,64,64ch] NHWC  67,108,864 (h* dead)
  float* pw2 = ws + 76546048;     // dc2 pack: 32*(32+32+32+64)   = 5,120
  float* pw3 = ws + 76551168;     // dc3 pack: 64*(32+64+64+128)  = 18,432
  float* pw4 = ws + 76569600;     // dc4 pack: 16*(64+128+128+256)= 9,216

  const int TB = 256;
  auto blocks = [](long total) { return (int)((total + 255) / 256); };

  // ---- Weight repacks for WMMA deconvs (tiny) ----
  pack_deconv_w<<<blocks(5120),  TB, 0, stream>>>(dc2_w, pw2, 16, 32, 32);
  pack_deconv_w<<<blocks(18432), TB, 0, stream>>>(dc3_w, pw3, 32, 64, 64);
  pack_deconv_w<<<blocks(9216),  TB, 0, stream>>>(dc4_w, pw4, 64, 3, 16);

  // ---- Encoder (bandwidth-bound direct convs, fused ReLU) ----
  conv2d_s2_relu<<<blocks((long)B * 16 * 64 * 64), TB, 0, stream>>>(
      x, conv1_w, conv1_b, a1, B, 3, 128, 128, 16, 64, 64);
  conv2d_s2_relu<<<blocks((long)B * 8 * 32 * 32), TB, 0, stream>>>(
      a1, conv2_w, conv2_b, a2, B, 16, 64, 64, 8, 32, 32);
  conv2d_s2_relu<<<blocks((long)B * 4 * 16 * 16), TB, 0, stream>>>(
      a2, conv3_w, conv3_b, a3, B, 8, 32, 32, 4, 16, 16);
  conv2d_s2_relu<<<blocks((long)B * 2 * 8 * 8), TB, 0, stream>>>(
      a3, conv4_w, conv4_b, a4, B, 4, 16, 16, 2, 8, 8);

  // ---- FC stack: v_wmma_f32_16x16x32_f16, one wave per 16x16 tile ----
  gemm_wmma<1><<<(256 / 16) * (256 / 16), 32, 0, stream>>>(a4, l2_w, l2_b, h1, 256, 256, 128);
  custom_layer_wmma<<<(256 / 16) * (512 / 16), 32, 0, stream>>>(
      h1, cl_w, cl_b, n_w, n_b, low, high, h2, 256, 512, 256);
  gemm_wmma<1><<<(256 / 16) * (256 / 16), 32, 0, stream>>>(h2, l4_w, l4_b, h3, 256, 256, 512);
  gemm_wmma<2><<<(256 / 16) * (256 / 16), 32, 0, stream>>>(h3, lL_w, lL_b, h4, 256, 256, 256);
  gemm_wmma<2><<<(256 / 16) * (512 / 16), 32, 0, stream>>>(h4, fc4_w, fc4_b, h5, 256, 512, 256);
  gemm_wmma<1><<<(256 / 16) * (512 / 16), 32, 0, stream>>>(h5, fc5_w, fc5_b, h6, 256, 512, 512);
  // h6 == [256,8,8,8] NCHW (row-major reshape).

  // ---- Decoder ----
  // dc1: tiny -> scalar, but emit NHWC for the WMMA layers downstream.
  deconv2d_s2<1><<<blocks((long)B * 16 * 16 * 16), TB, 0, stream>>>(
      h6, dc1_w, dc1_b, e1, B, 8, 8, 8, 16, 16, 16);
  // dc2: NHWC->NHWC WMMA implicit GEMM. 4 classes x (65536/16) x (32/16) waves.
  deconv_wmma<0><<<4 * ((B * 16 * 16) / 16) * (32 / 16), 32, 0, stream>>>(
      e1, pw2, dc2_b, e2, B, 16, 16, 16, 32, 32);
  // dc3: NHWC->NHWC WMMA implicit GEMM (dominant layer).
  deconv_wmma<0><<<4 * ((B * 32 * 32) / 16) * (64 / 16), 32, 0, stream>>>(
      e2, pw3, dc3_b, e3, B, 32, 32, 32, 64, 64);
  // dc4: NHWC->NCHW WMMA implicit GEMM, N padded 3->16 with masked stores.
  deconv_wmma<1><<<4 * ((B * 64 * 64) / 16) * (16 / 16), 32, 0, stream>>>(
      e3, pw4, dc4_b, out, B, 64, 64, 64, 3, 16);
}
